// SensorCoverageOptimizer_8143257993800
// MI455X (gfx1250) — matrix operand order repaired and checked by reading, not compiled, not run on hardware
//
#include <hip/hip_runtime.h>

typedef __attribute__((ext_vector_type(16))) _Float16 v16h;
typedef __attribute__((ext_vector_type(8)))  float    v8f;

#define NT     300          // targets
#define NS     30           // sensors
#define GRIDN  512
#define R2     2500.0f      // RADIUS^2   (exact in f32; sqrt-compare equivalent)
#define MOVE2  62500.0f     // MOVE_LIMIT^2
#define KCH    10           // K chunks of 32 -> Tpad = 320
#define TPAD   (KCH * 32)
#define BFRAG_ELEMS (KCH * 2 * 32 * 16)   // [kchunk][ntile][lane][16] halves

// ---------------------------------------------------------------------------
// Kernel 1: tiny setup. Computes per-target coverage counts/masks,
// current_reward, off[s] = base[s] - current_reward, and the B matrix
// M[t,s] = w_t * [no sensor other than s covers t], pre-swizzled into the
// per-lane WMMA B-fragment layout (16-bit B 32x16: lane = col N,
// element e -> K = (e<8 ? e : e+8) + 8*(lane>>4)).
// Deterministic: all reductions are sequential per-thread loops.
// ---------------------------------------------------------------------------
__global__ __launch_bounds__(320) void sensor_prep_kernel(
    const float* __restrict__ tx, const float* __restrict__ ty,
    const float* __restrict__ tw, const float* __restrict__ sx,
    const float* __restrict__ sy,
    float* __restrict__ off_out,          // [32] floats in ws
    _Float16* __restrict__ bfrag_out)     // [BFRAG_ELEMS] halves in ws
{
#pragma clang fp contract(off)            // match reference rounding exactly
    __shared__ int      s_cnt[NT];
    __shared__ unsigned s_mask[NT];

    const int tid = threadIdx.x;

    if (tid < NT) {
        const float x = tx[tid], y = ty[tid];
        unsigned m = 0u; int c = 0;
        for (int s = 0; s < NS; ++s) {
            const float dx = sx[s] - x, dy = sy[s] - y;
            const float d2 = dx * dx + dy * dy;     // separate roundings
            if (d2 <= R2) { m |= (1u << s); ++c; }
        }
        s_cnt[tid]  = c;
        s_mask[tid] = m;
    }
    __syncthreads();

    if (tid < 32) {
        float off = 0.0f;
        if (tid < NS) {
            // current_reward + base[s], deterministic sequential sums
            float cur = 0.0f, base = 0.0f;
            for (int t = 0; t < NT; ++t) {
                const float w = tw[t];
                if (s_cnt[t] > 0) cur += w;
                const int covst = (int)((s_mask[t] >> tid) & 1u);
                if (s_cnt[t] - covst > 0) base += w;   // covered by others
            }
            off = base - cur;
        }
        off_out[tid] = off;
    }

    // B fragments, swizzled for v_wmma_f32_16x16x32_f16 B operand.
    for (int idx = tid; idx < BFRAG_ELEMS; idx += 320) {
        const int e     = idx & 15;
        const int lane  = (idx >> 4) & 31;
        const int ntile = (idx >> 9) & 1;
        const int k     = idx >> 10;
        const int t     = k * 32 + (e < 8 ? e : e + 8) + 8 * (lane >> 4);   // B: K = (e<8 ? e : e+8) + 8*hi
        const int s     = ntile * 16 + (lane & 15);        // B: N = lane&15
        float v = 0.0f;
        if (t < NT && s < NS) {
            const int covst = (int)((s_mask[t] >> s) & 1u);
            if (s_cnt[t] - covst == 0) v = tw[t];          // not covered by others
        }
        bfrag_out[idx] = (_Float16)v;
    }
}

// ---------------------------------------------------------------------------
// Kernel 2: fused covers-generation + GEMM + epilogue.
// 256 threads = 8 waves; each wave owns 16 consecutive cells of one grid row
// (block owns 128 cells, row-aligned -> g uniform). Per-block precompute:
// (g - tx[t])^2 hoisted into LDS (uniform over the block), so the hot
// A-fragment loop is just {sub, mul, add, cmp, sel} per element.
// K-loop: 10 chunks of 32 targets, 2 WMMAs per chunk (sensor tiles 0-15 /
// 16-31), f32 accumulation (exact: products are small integers).
// Accumulators transposed through padded LDS for coalesced stores.
// ---------------------------------------------------------------------------
__global__ __launch_bounds__(256) void sensor_main_kernel(
    const float* __restrict__ tx, const float* __restrict__ ty,
    const float* __restrict__ sx, const float* __restrict__ sy,
    const float* __restrict__ off_in,
    const _Float16* __restrict__ bfrag_in,
    float* __restrict__ out)
{
#pragma clang fp contract(off)            // match reference rounding exactly
    __shared__ float s_prep[TPAD * 2];                // packed (dx^2, ty), 2560 B
    __shared__ v16h  s_bfrag[KCH * 2 * 32];           // 20480 B, 32B-aligned
    __shared__ float s_stage[32 * 133];               // padded transpose buffer

    const int tid = threadIdx.x;

    const int  block_cell = blockIdx.x * 128;   // 128 cells, row-aligned
    const int  g  = block_cell >> 9;            // uniform grid row
    const int  h0 = block_cell & (GRIDN - 1);   // aligned to 128
    const float gf = (float)g;

    // Stage (dx^2, ty) pairs; padded targets pushed far away -> never covered.
    for (int t = tid; t < TPAD; t += 256) {
        float x = 1.0e9f, y = 1.0e9f;
        if (t < NT) { x = tx[t]; y = ty[t]; }
        const float dx = gf - x;                // reference: g - target_x
        s_prep[2 * t]     = dx * dx;            // rounded square, as reference
        s_prep[2 * t + 1] = y;
    }
    // Stage swizzled B fragments.
    {
        _Float16* sb = (_Float16*)s_bfrag;
        for (int i = tid; i < BFRAG_ELEMS; i += 256) sb[i] = bfrag_in[i];
    }
    __syncthreads();

    const int lane = tid & 31;
    const int wave = tid >> 5;
    const int hi   = lane >> 4;                 // half-wave group
    const int rowM = lane & 15;                 // A-matrix row within tile

    const float hfA = (float)(h0 + wave * 16 + rowM);  // this lane's A-row cell

    v8f acc0 = {};
    v8f acc1 = {};

    for (int k = 0; k < KCH; ++k) {
        // Build A fragment: covers for 16 cells x 32 targets (this chunk).
        // A layout: K = (e<8 ? e : e+8) + 8*hi.
        v16h a;
        const int tbase = k * 32 + hi * 8;
#pragma unroll
        for (int e = 0; e < 16; ++e) {
            const int   t   = tbase + (e < 8 ? e : e + 8);
            const float dx2 = s_prep[2 * t];
            const float y   = s_prep[2 * t + 1];
            const float dy  = hfA - y;          // reference: h - target_y
            const float d2  = dx2 + dy * dy;    // separate roundings
            a[e] = (d2 <= R2) ? (_Float16)1.0f : (_Float16)0.0f;
        }
        const v16h b0 = s_bfrag[(k * 2 + 0) * 32 + lane];
        const v16h b1 = s_bfrag[(k * 2 + 1) * 32 + lane];
        acc0 = __builtin_amdgcn_wmma_f32_16x16x32_f16(
            false, a, false, b0, (short)0, acc0, false, false);
        acc1 = __builtin_amdgcn_wmma_f32_16x16x32_f16(
            false, a, false, b1, (short)0, acc1, false, false);
    }

    // Transpose through LDS. D layout: element r -> cell m = r + 8*hi,
    // sensor n = lane&15 (+16 for acc1). Stride 133 -> conflict-free.
    {
        const int n0 = lane & 15;
#pragma unroll
        for (int r = 0; r < 8; ++r) {
            const int c = wave * 16 + r + 8 * hi;   // cell within block
            s_stage[n0 * 133 + c]        = acc0[r];
            s_stage[(16 + n0) * 133 + c] = acc1[r];
        }
    }
    __syncthreads();

    // Epilogue + coalesced stores: thread-half 0 does even sensors, half 1 odd.
    const int   c    = tid & 127;
    const int   cell = block_cell + c;
    const float hcf  = (float)(h0 + c);
    for (int s = (tid >> 7); s < NS; s += 2) {
        const float rx = sx[s] - gf;            // reach: sensor - grid index
        const float ry = sy[s] - hcf;
        const float r2 = rx * rx + ry * ry;     // separate roundings
        float v = s_stage[s * 133 + c] + off_in[s];   // reward - current_reward
        v = v > 0.0f ? v : 0.0f;
        v = (r2 <= MOVE2) ? v : 0.0f;
        out[s * (GRIDN * GRIDN) + cell] = v;
    }
}

extern "C" void kernel_launch(void* const* d_in, const int* in_sizes, int n_in,
                              void* d_out, int out_size, void* d_ws, size_t ws_size,
                              hipStream_t stream) {
    const float* tx = (const float*)d_in[0];
    const float* ty = (const float*)d_in[1];
    const float* tw = (const float*)d_in[2];
    const float* sx = (const float*)d_in[3];
    const float* sy = (const float*)d_in[4];
    float* out = (float*)d_out;

    float*    off   = (float*)d_ws;                      // 32 floats
    _Float16* bfrag = (_Float16*)((char*)d_ws + 128);    // 20480 bytes

    sensor_prep_kernel<<<1, 320, 0, stream>>>(tx, ty, tw, sx, sy, off, bfrag);

    const int nblocks = (GRIDN * GRIDN) / 128;           // 2048
    sensor_main_kernel<<<nblocks, 256, 0, stream>>>(tx, ty, sx, sy, off, bfrag, out);
}
